// AttentionV2_88914412962153
// MI455X (gfx1250) — matrix operand order repaired and checked by reading, not compiled, not run on hardware
//
#include <hip/hip_runtime.h>

#define BB    32
#define CC    512
#define HEADS 8
#define DD    64
#define NN    4096  // H*W

typedef __attribute__((ext_vector_type(16))) __bf16 v16bf;
typedef __attribute__((ext_vector_type(8)))  float  v8f;
typedef __attribute__((ext_vector_type(4)))  unsigned v4u;
typedef __attribute__((ext_vector_type(8)))  unsigned v8u;

union Frag16 { v16bf v; unsigned u[8]; };

__device__ __forceinline__ unsigned short f2bf(float f) {
    unsigned u = __float_as_uint(f);
    u += 0x7fffu + ((u >> 16) & 1u);   // round-to-nearest-even
    return (unsigned short)(u >> 16);
}

// ---------------------------------------------------------------------------
// K1: per (b,h): G = Qraw * Kraw^T  (64x64, K=4096) + row L2 norms (fused).
// grid = B*HEADS blocks, 256 threads (8 waves). bf16 WMMA, fp32 accumulate.
// ---------------------------------------------------------------------------
__global__ __launch_bounds__(256)
void ca_k1_gram_norm(const float* __restrict__ loc, const float* __restrict__ glb,
                     float* __restrict__ G, float* __restrict__ NQ,
                     float* __restrict__ NK)
{
    __shared__ __align__(16) unsigned short qt[DD * 72];
    __shared__ __align__(16) unsigned short kt[DD * 72];
    __shared__ float red[256];

    const int blk  = blockIdx.x;           // b*HEADS + h
    const int t    = threadIdx.x;
    const int lane = t & 31;
    const int w    = t >> 5;
    const int g    = lane >> 4;
    const int lm   = lane & 15;

    const int r  = t >> 2;                 // row 0..63 handled for loads
    const int qc = (t & 3) * 16;           // 16-col quarter of the 64-chunk

    const float* qrow = loc + ((size_t)blk * DD + r) * NN;
    const float* krow = glb + ((size_t)blk * DD + r) * NN;

    float qss = 0.f, kss = 0.f;

    const int tm  = w >> 1;                // output tile row 0..3
    const int tn0 = (w & 1) * 2;           // output tile cols tn0, tn0+1
    v8f acc0 = {};
    v8f acc1 = {};

    for (int kb = 0; kb < NN; kb += 64) {
        const float4* q4 = (const float4*)(qrow + kb + qc);
        const float4* k4 = (const float4*)(krow + kb + qc);
        #pragma unroll
        for (int i = 0; i < 4; ++i) {
            float4 a = q4[i];
            float4 b = k4[i];
            qss += a.x*a.x + a.y*a.y + a.z*a.z + a.w*a.w;
            kss += b.x*b.x + b.y*b.y + b.z*b.z + b.w*b.w;
            const int c0 = qc + i * 4;
            qt[r*72 + c0+0] = f2bf(a.x); qt[r*72 + c0+1] = f2bf(a.y);
            qt[r*72 + c0+2] = f2bf(a.z); qt[r*72 + c0+3] = f2bf(a.w);
            kt[r*72 + c0+0] = f2bf(b.x); kt[r*72 + c0+1] = f2bf(b.y);
            kt[r*72 + c0+2] = f2bf(b.z); kt[r*72 + c0+3] = f2bf(b.w);
        }
        __syncthreads();

        #pragma unroll
        for (int kk = 0; kk < 64; kk += 32) {
            Frag16 fa, fb0, fb1;
            const int am = tm * 16 + lm;
            #pragma unroll
            for (int p = 0; p < 8; ++p) {
                const int kidx = kk + g * 16 + 2 * p;
                fa.u[p] = *(const unsigned*)&qt[am * 72 + kidx];
            }
            const int bn0 = tn0 * 16 + lm;
            const int kb2 = kk + g * 16;
            #pragma unroll
            for (int p = 0; p < 8; ++p) {
                fb0.u[p] = *(const unsigned*)&kt[bn0 * 72 + kb2 + 2 * p];
                fb1.u[p] = *(const unsigned*)&kt[(bn0 + 16) * 72 + kb2 + 2 * p];
            }
            acc0 = __builtin_amdgcn_wmma_f32_16x16x32_bf16(false, fa.v, false, fb0.v,
                                                           (short)0, acc0, false, false);
            acc1 = __builtin_amdgcn_wmma_f32_16x16x32_bf16(false, fa.v, false, fb1.v,
                                                           (short)0, acc1, false, false);
        }
        __syncthreads();
    }

    // Write G tiles (C/D layout: VGPR rr -> rows rr / rr+8, col = lane&15)
    {
        float* Gb = G + (size_t)blk * DD * DD;
        const int col0   = tn0 * 16 + lm;
        const int rowoff = g * 8;
        #pragma unroll
        for (int rr = 0; rr < 8; ++rr) {
            const int m = tm * 16 + rowoff + rr;
            Gb[m * DD + col0]      = acc0[rr];
            Gb[m * DD + col0 + 16] = acc1[rr];
        }
    }

    // Row-norm reductions (4 partials per row)
    red[t] = qss;
    __syncthreads();
    if (t < 64) {
        float s = red[t*4] + red[t*4+1] + red[t*4+2] + red[t*4+3];
        NQ[blk * DD + t] = sqrtf(s);
    }
    __syncthreads();
    red[t] = kss;
    __syncthreads();
    if (t < 64) {
        float s = red[t*4] + red[t*4+1] + red[t*4+2] + red[t*4+3];
        NK[blk * DD + t] = sqrtf(s);
    }
}

// ---------------------------------------------------------------------------
// K2: softmax over d of G[c,d]/(max(nq,eps)*max(nk,eps))*temp  -> bf16 A.
// grid = B*HEADS, 64 threads (one row per thread; data is L2-hot, 4 MB total)
// ---------------------------------------------------------------------------
__global__ __launch_bounds__(64)
void ca_k2_softmax(const float* __restrict__ G, const float* __restrict__ NQ,
                   const float* __restrict__ NK, const float* __restrict__ temp,
                   unsigned short* __restrict__ Abf)
{
    const int blk = blockIdx.x;
    const int h   = blk & (HEADS - 1);
    const int r   = threadIdx.x;

    const float* Gr  = G  + ((size_t)blk * DD + r) * DD;
    const float* nkb = NK + (size_t)blk * DD;
    const float  rq  = temp[h] / fmaxf(NQ[blk * DD + r], 1e-12f);

    float mx = -1e30f;
    for (int d = 0; d < DD; ++d) {
        float s = Gr[d] * rq / fmaxf(nkb[d], 1e-12f);
        mx = fmaxf(mx, s);
    }
    float sum = 0.f;
    for (int d = 0; d < DD; ++d) {
        float s = Gr[d] * rq / fmaxf(nkb[d], 1e-12f);
        sum += __expf(s - mx);
    }
    const float rs = 1.f / sum;
    unsigned short* Ar = Abf + ((size_t)blk * DD + r) * DD;
    for (int d = 0; d < DD; ++d) {
        float s = Gr[d] * rq / fmaxf(nkb[d], 1e-12f);
        Ar[d] = f2bf(__expf(s - mx) * rs);
    }
}

// ---------------------------------------------------------------------------
// K3: M[b][o][h*64+d] = sum_j W[o, h*64+j] * A[b,h][j][d]   (WMMA)
// grid = B*HEADS*2 (half of the 512 o-rows each), 256 threads.
// ---------------------------------------------------------------------------
__global__ __launch_bounds__(256)
void ca_k3_formM(const float* __restrict__ Wp, const unsigned short* __restrict__ Abf,
                 unsigned short* __restrict__ Mbf)
{
    __shared__ __align__(16) unsigned short wt[256 * 72];  // [o_local][j]
    __shared__ __align__(16) unsigned short att[DD * 72];  // [d][j]  (A transposed)

    const int blk  = blockIdx.x;
    const int b    = blk >> 4;
    const int rem  = blk & 15;
    const int h    = rem >> 1;
    const int half = rem & 1;
    const int obase = half * 256;

    const int t    = threadIdx.x;
    const int lane = t & 31;
    const int w    = t >> 5;
    const int g    = lane >> 4;
    const int lm   = lane & 15;

    for (int idx = t; idx < 256 * 64; idx += 256) {
        const int o = idx >> 6, j = idx & 63;
        wt[o * 72 + j] = f2bf(Wp[(size_t)(obase + o) * CC + h * DD + j]);
    }
    const unsigned short* Ab = Abf + ((size_t)(b * HEADS + h)) * DD * DD;
    for (int idx = t; idx < DD * DD; idx += 256) {
        const int rr = idx >> 6, d = idx & 63;
        att[d * 72 + rr] = Ab[idx];
    }
    __syncthreads();

    v8f acc[2][4] = {};
    #pragma unroll
    for (int kk = 0; kk < 64; kk += 32) {
        #pragma unroll
        for (int tr2 = 0; tr2 < 2; ++tr2) {
            Frag16 fa;
            const int m = (w * 2 + tr2) * 16 + lm;
            #pragma unroll
            for (int p = 0; p < 8; ++p) {
                const int kidx = kk + g * 16 + 2 * p;
                fa.u[p] = *(const unsigned*)&wt[m * 72 + kidx];
            }
            const int kb2 = kk + g * 16;
            #pragma unroll
            for (int tc = 0; tc < 4; ++tc) {
                Frag16 fb;
                const int n = tc * 16 + lm;
                #pragma unroll
                for (int p = 0; p < 8; ++p)
                    fb.u[p] = *(const unsigned*)&att[n * 72 + kb2 + 2 * p];
                acc[tr2][tc] = __builtin_amdgcn_wmma_f32_16x16x32_bf16(
                    false, fa.v, false, fb.v, (short)0, acc[tr2][tc], false, false);
            }
        }
    }

    unsigned short* Mb = Mbf + ((size_t)b * CC + obase) * CC + h * DD;
    const int rowoff = g * 8;
    #pragma unroll
    for (int tr2 = 0; tr2 < 2; ++tr2)
        #pragma unroll
        for (int tc = 0; tc < 4; ++tc)
            #pragma unroll
            for (int rr = 0; rr < 8; ++rr) {
                const int m = (w * 2 + tr2) * 16 + rowoff + rr;
                const int n = tc * 16 + lm;
                Mb[(size_t)m * CC + n] = f2bf(acc[tr2][tc][rr]);
            }
}

// ---------------------------------------------------------------------------
// K4: final[b] = M[b] (512x512 bf16) @ V[b] (512x4096 fp32->bf16), fp32 out.
// grid = (4096/128, 512/128, B), 256 threads; 128x128 tile per block.
// M tile (bf16, no conversion needed) is fetched by the Tensor Data Mover
// (tensor_load_to_lds) with hardware LDS row padding (128B rows + 16B pad ->
// 144B = 72-bf16 stride); V tile is converted fp32->bf16 on VALU in parallel.
// ---------------------------------------------------------------------------
__global__ __launch_bounds__(256)
void ca_k4_gemm(const unsigned short* __restrict__ Mbf, const float* __restrict__ glb,
                float* __restrict__ out)
{
    __shared__ __align__(16) unsigned short mt[128 * 72];  // [m][k]  (TDM-filled)
    __shared__ __align__(16) unsigned short vt[128 * 72];  // [n][k]  (V transposed)

    const int nb = blockIdx.x * 128;
    const int mb = blockIdx.y * 128;
    const int b  = blockIdx.z;

    const int t    = threadIdx.x;
    const int lane = t & 31;
    const int w    = t >> 5;
    const int g    = lane >> 4;
    const int lm   = lane & 15;
    const int wm   = w >> 2;   // 0..1 : 64-row half
    const int wn   = w & 3;    // 0..3 : 32-col slice

    const unsigned ldsbase = (unsigned)(size_t)(void*)&mt[0]; // LDS byte offset

    v8f acc[4][2] = {};

    for (int kb = 0; kb < CC; kb += 64) {
        if (w == 0) {
            // Tensor DMA: 2D tile, 128 rows x 64 bf16, row pitch 512 elements,
            // LDS pad: after every 32 DWORDs (one 128B row) insert 4 DWORDs.
            const unsigned long long ga =
                (unsigned long long)(const void*)(Mbf + ((size_t)b * CC + mb) * CC + kb);
            v4u g0;
            g0[0] = 1u;                                            // count=1
            g0[1] = ldsbase;                                       // lds_addr
            g0[2] = (unsigned)ga;                                  // global_addr lo
            g0[3] = ((unsigned)(ga >> 32) & 0x01FFFFFFu) | (2u << 30); // hi | type=2
            v8u g1;
            g1[0] = (1u << 16)      // data_size = 2 bytes
                  | (1u << 20)      // pad_enable
                  | (4u << 22)      // pad_interval: 32 DWORDs
                  | (3u << 25);     // pad_amount : 4 DWORDs
            g1[1] = (64u  << 16);   // tensor_dim0 = 64  (bits 79:48, lo16)
            g1[2] = (128u << 16);   // tensor_dim0 hi=0 | tensor_dim1 = 128 (lo16)
            g1[3] = (64u  << 16);   // tensor_dim1 hi=0 | tile_dim0 = 64
            g1[4] = 128u;           // tile_dim1 = 128 | tile_dim2 = 0
            g1[5] = 512u;           // tensor_dim0_stride = 512 elements (lo32)
            g1[6] = 0u;             // stride hi | tensor_dim1_stride lo
            g1[7] = 0u;
            asm volatile("tensor_load_to_lds %0, %1" :: "s"(g0), "s"(g1) : "memory");
        }

        // V tile: 64 x 128 fp32 -> bf16, transposed into vt[n][k]
        for (int idx = t; idx < 64 * 128; idx += 256) {
            const int kr = idx >> 7, nc = idx & 127;
            vt[nc * 72 + kr] = f2bf(glb[((size_t)b * CC + kb + kr) * NN + nb + nc]);
        }

        if (w == 0) __builtin_amdgcn_s_wait_tensorcnt(0);
        __syncthreads();

        #pragma unroll
        for (int kk = 0; kk < 64; kk += 32) {
            Frag16 fa[4];
            #pragma unroll
            for (int i = 0; i < 4; ++i) {
                const int m = wm * 64 + i * 16 + lm;
                #pragma unroll
                for (int p = 0; p < 8; ++p) {
                    const int kidx = kk + g * 16 + 2 * p;
                    fa[i].u[p] = *(const unsigned*)&mt[m * 72 + kidx];
                }
            }
            const int kb2 = kk + g * 16;
            #pragma unroll
            for (int j = 0; j < 2; ++j) {
                Frag16 fb;
                const int n = wn * 32 + j * 16 + lm;
                #pragma unroll
                for (int p = 0; p < 8; ++p)
                    fb.u[p] = *(const unsigned*)&vt[n * 72 + kb2 + 2 * p];
                #pragma unroll
                for (int i = 0; i < 4; ++i)
                    acc[i][j] = __builtin_amdgcn_wmma_f32_16x16x32_bf16(
                        false, fa[i].v, false, fb.v, (short)0, acc[i][j], false, false);
            }
        }
        __syncthreads();
    }

    const int rowoff = g * 8;
    #pragma unroll
    for (int i = 0; i < 4; ++i)
        #pragma unroll
        for (int j = 0; j < 2; ++j)
            #pragma unroll
            for (int rr = 0; rr < 8; ++rr) {
                const int m = mb + wm * 64 + i * 16 + rowoff + rr;
                const int n = nb + wn * 32 + j * 16 + lm;
                out[((size_t)b * CC + m) * NN + n] = acc[i][j][rr];
            }
}

// ---------------------------------------------------------------------------
extern "C" void kernel_launch(void* const* d_in, const int* in_sizes, int n_in,
                              void* d_out, int out_size, void* d_ws, size_t ws_size,
                              hipStream_t stream)
{
    (void)in_sizes; (void)n_in; (void)out_size; (void)ws_size;

    const float* loc  = (const float*)d_in[0];  // local_feature  [32,512,64,64]
    const float* glb  = (const float*)d_in[1];  // global_feature [32,512,64,64]
    const float* temp = (const float*)d_in[2];  // temperature [8]
    const float* pw   = (const float*)d_in[3];  // proj_w [512,512]
    float* out        = (float*)d_out;

    char* ws = (char*)d_ws;
    float*          G   = (float*)(ws + 0);                 //  4 MB  (B*H*64*64 f32)
    float*          NQ  = (float*)(ws + 4194304);           // 64 KB
    float*          NK  = (float*)(ws + 4259840);           // 64 KB
    unsigned short* Abf = (unsigned short*)(ws + 4325376);  //  2 MB  (bf16 attn)
    unsigned short* Mbf = (unsigned short*)(ws + 6422528);  // 16 MB  (bf16 fused M)

    ca_k1_gram_norm<<<BB * HEADS, 256, 0, stream>>>(loc, glb, G, NQ, NK);
    ca_k2_softmax <<<BB * HEADS,  64, 0, stream>>>(G, NQ, NK, temp, Abf);
    ca_k3_formM   <<<BB * HEADS * 2, 256, 0, stream>>>(pw, Abf, Mbf);
    ca_k4_gemm    <<<dim3(NN / 128, CC / 128, BB), 256, 0, stream>>>(Mbf, glb, out);
}
